// MyModelTorch_266287972381
// MI455X (gfx1250) — matrix operand order repaired and checked by reading, not compiled, not run on hardware
//
#include <hip/hip_runtime.h>

typedef __attribute__((ext_vector_type(16))) _Float16 v16h;
typedef __attribute__((ext_vector_type(8)))  _Float16 v8h;
typedef __attribute__((ext_vector_type(8)))  float    v8f;

#define NHEADS 4
#define HD     32
#define SEQ    144
#define NT     9        // SEQ / 16
#define KPAD   160      // SEQ padded to multiple of 32 for the AV K-loop
#define C3     384
#define ATT_SCALE 0.17677669529663687f

union V16U { v16h v; v8h h[2]; };

__global__ __launch_bounds__(288)
void attn_fused_kernel(const float* __restrict__ qkv,
                       const float* __restrict__ mask,
                       float* __restrict__ out)
{
    __shared__ __align__(32) _Float16 K_lds[SEQ * HD];        //  9216 B
    __shared__ __align__(32) _Float16 Vt_lds[HD * KPAD];      // 10240 B, V transposed + zero pad
    __shared__ __align__(32) _Float16 P_lds[NT][16 * SEQ];    // 41472 B, per-wave slabs

    const int b    = blockIdx.x >> 2;
    const int h    = blockIdx.x & 3;
    const int tid  = threadIdx.x;
    const int wave = tid >> 5;     // Q row-block index, 0..8
    const int lane = tid & 31;
    const int r    = lane & 15;    // A: row / B: column / C: column index
    const int s    = lane >> 4;    // lane half

    // ---- stage K (f16) and V^T (f16, zero-padded) into LDS; coalesced global reads ----
    const size_t bbase = (size_t)b * SEQ * C3 + (size_t)h * HD;
    for (int idx = tid; idx < SEQ * HD; idx += 288) {
        const int n = idx >> 5;
        const int d = idx & 31;
        const size_t g = bbase + (size_t)n * C3 + d;
        K_lds[idx]           = (_Float16)qkv[g + 128];   // K at t=1
        Vt_lds[d * KPAD + n] = (_Float16)qkv[g + 256];   // V at t=2, transposed
    }
    for (int idx = tid; idx < HD * (KPAD - SEQ); idx += 288) {
        const int d = idx >> 4;
        Vt_lds[d * KPAD + SEQ + (idx & 15)] = (_Float16)0.f;
    }
    __syncthreads();

    // ---- Q tile as A operand (16x32 f16, pre-scaled), lane r = row, halves pack K dims ----
    v16h qa;
    {
        const float* qp = qkv + bbase + (size_t)(wave * 16 + r) * C3;
        const float4 a0 = *(const float4*)(qp + 16 * s);
        const float4 a1 = *(const float4*)(qp + 16 * s + 4);
        const float4 a2 = *(const float4*)(qp + 16 * s + 8);
        const float4 a3 = *(const float4*)(qp + 16 * s + 12);
        qa[0]  = (_Float16)(a0.x * ATT_SCALE); qa[1]  = (_Float16)(a0.y * ATT_SCALE);
        qa[2]  = (_Float16)(a0.z * ATT_SCALE); qa[3]  = (_Float16)(a0.w * ATT_SCALE);
        qa[4]  = (_Float16)(a1.x * ATT_SCALE); qa[5]  = (_Float16)(a1.y * ATT_SCALE);
        qa[6]  = (_Float16)(a1.z * ATT_SCALE); qa[7]  = (_Float16)(a1.w * ATT_SCALE);
        qa[8]  = (_Float16)(a2.x * ATT_SCALE); qa[9]  = (_Float16)(a2.y * ATT_SCALE);
        qa[10] = (_Float16)(a2.z * ATT_SCALE); qa[11] = (_Float16)(a2.w * ATT_SCALE);
        qa[12] = (_Float16)(a3.x * ATT_SCALE); qa[13] = (_Float16)(a3.y * ATT_SCALE);
        qa[14] = (_Float16)(a3.z * ATT_SCALE); qa[15] = (_Float16)(a3.w * ATT_SCALE);
    }

    // ---- scores S = Q·K^T + mask : 9 WMMAs, C layout: lane=col, VGPR v=row(v+8s) ----
    v8f sc[NT];
    float mx[8];
#pragma unroll
    for (int v = 0; v < 8; ++v) mx[v] = -3.0e38f;

    const float* mbase = mask + ((size_t)h * SEQ + wave * 16 + 8 * s) * SEQ + r;

#pragma unroll
    for (int j = 0; j < NT; ++j) {
        // B operand: lane r = key row n, contiguous 16 K-dims per lane half
        const v16h kb = *(const v16h*)&K_lds[(j * 16 + r) * HD + 16 * s];
        v8f c = {};
        c = __builtin_amdgcn_wmma_f32_16x16x32_f16(false, qa, false, kb,
                                                   (short)0, c, false, false);
#pragma unroll
        for (int v = 0; v < 8; ++v) {
            c[v] += mbase[v * SEQ + j * 16];
            mx[v] = fmaxf(mx[v], c[v]);
        }
        sc[j] = c;
    }

    // ---- softmax stats: reduce across the 16 lanes holding each row ----
#pragma unroll
    for (int v = 0; v < 8; ++v) {
        float m = mx[v];
        m = fmaxf(m, __shfl_xor(m, 1));
        m = fmaxf(m, __shfl_xor(m, 2));
        m = fmaxf(m, __shfl_xor(m, 4));
        m = fmaxf(m, __shfl_xor(m, 8));
        mx[v] = m;
    }
    float sm[8];
#pragma unroll
    for (int v = 0; v < 8; ++v) sm[v] = 0.f;
#pragma unroll
    for (int j = 0; j < NT; ++j) {
#pragma unroll
        for (int v = 0; v < 8; ++v) {
            const float e = __expf(sc[j][v] - mx[v]);
            sc[j][v] = e;
            sm[v] += e;
        }
    }
#pragma unroll
    for (int v = 0; v < 8; ++v) {
        float t = sm[v];
        t += __shfl_xor(t, 1);
        t += __shfl_xor(t, 2);
        t += __shfl_xor(t, 4);
        t += __shfl_xor(t, 8);
        sm[v] = t;
    }

    // ---- stage unnormalized P (f16) into this wave's private slab (C->A relayout) ----
    _Float16* pslab = &P_lds[wave][0];
#pragma unroll
    for (int j = 0; j < NT; ++j) {
#pragma unroll
        for (int v = 0; v < 8; ++v) {
            pslab[(v + 8 * s) * SEQ + j * 16 + r] = (_Float16)sc[j][v];
        }
    }
    // same-wave DS ops are in order; no barrier needed (slab is wave-private)

    // ---- O = P·V : K padded to 160, 5 chunks x 2 column halves = 10 WMMAs ----
    v8f acc0 = {}, acc1 = {};
    const _Float16* prow = &P_lds[wave][r * SEQ];
#pragma unroll
    for (int kk = 0; kk < KPAD; kk += 32) {
        V16U pa;
        if (kk + 16 * s < SEQ) {
            pa.h[0] = *(const v8h*)(prow + kk + 16 * s);       // k = kk+16s .. +7
            pa.h[1] = *(const v8h*)(prow + kk + 16 * s + 8);   // k = kk+16s+8 .. +7
        } else {
            v8h z = {};
            pa.h[0] = z; pa.h[1] = z;                          // zero tail (k >= 144)
        }
        const v16h vb0 = *(const v16h*)&Vt_lds[r * KPAD + kk + 16 * s];
        const v16h vb1 = *(const v16h*)&Vt_lds[(16 + r) * KPAD + kk + 16 * s];
        acc0 = __builtin_amdgcn_wmma_f32_16x16x32_f16(false, pa.v, false, vb0,
                                                      (short)0, acc0, false, false);
        acc1 = __builtin_amdgcn_wmma_f32_16x16x32_f16(false, pa.v, false, vb1,
                                                      (short)0, acc1, false, false);
    }

    // ---- normalize by row sum (same row->VGPR mapping as D tile) and store ----
    float* obase = out + ((size_t)b * SEQ + wave * 16 + 8 * s) * (C3 / 3) + h * HD + r;
#pragma unroll
    for (int v = 0; v < 8; ++v) {
        const float inv = 1.0f / sm[v];
        obase[(size_t)v * (C3 / 3)]      = acc0[v] * inv;
        obase[(size_t)v * (C3 / 3) + 16] = acc1[v] * inv;
    }
}

extern "C" void kernel_launch(void* const* d_in, const int* in_sizes, int n_in,
                              void* d_out, int out_size, void* d_ws, size_t ws_size,
                              hipStream_t stream) {
    const float* qkv  = (const float*)d_in[0];
    const float* mask = (const float*)d_in[1];
    float* out = (float*)d_out;
    const int B = in_sizes[0] / (SEQ * C3);   // 1024
    attn_fused_kernel<<<dim3(B * NHEADS), dim3(288), 0, stream>>>(qkv, mask, out);
}
